// FNO_26371099198003
// MI455X (gfx1250) — matrix operand order repaired and checked by reading, not compiled, not run on hardware
//
#include <hip/hip_runtime.h>
#include <math.h>

// ---------------- problem constants ----------------
#define NB   16            // batch
#define NC   64            // channels N1 == N2
#define NH   256
#define NW   256
#define NK   12            // truncated modes
#define NL   4             // layers
#define HWP  (NH*NW)       // 65536 pixels
#define NR   (NB*NC*NH)    // 262144 "rows" for the W-axis GEMMs
#define NBC  (NB*NC)       // 1024
#define PI_F 3.14159265358979323846f

typedef __bf16 bf16;
typedef __attribute__((ext_vector_type(16))) __bf16 v16bf;
typedef __attribute__((ext_vector_type(8)))  float  v8f;

// round-to-nearest-even f32 -> bf16
__device__ __forceinline__ bf16 f2bf(float f) {
    unsigned u = __builtin_bit_cast(unsigned, f);
    unsigned r = (u + 0x7FFFu + ((u >> 16) & 1u)) >> 16;
    unsigned short s = (unsigned short)r;
    return __builtin_bit_cast(bf16, s);
}
__device__ __forceinline__ float bf2f(bf16 x) {
    unsigned u = ((unsigned)__builtin_bit_cast(unsigned short, x)) << 16;
    return __builtin_bit_cast(float, u);
}
__device__ __forceinline__ unsigned pack2f(float lo, float hi) {
    unsigned a = (unsigned)__builtin_bit_cast(unsigned short, f2bf(lo));
    unsigned b = (unsigned)__builtin_bit_cast(unsigned short, f2bf(hi));
    return a | (b << 16);
}
__device__ __forceinline__ unsigned pack2b(bf16 lo, bf16 hi) {
    unsigned a = (unsigned)__builtin_bit_cast(unsigned short, lo);
    unsigned b = (unsigned)__builtin_bit_cast(unsigned short, hi);
    return a | (b << 16);
}

__device__ __forceinline__ v8f wmma_bf16(v16bf a, v16bf b, v8f c) {
    return __builtin_amdgcn_wmma_f32_16x16x32_bf16(
        false, a, false, b, (short)0, c, false, false);
}

// Fragment-order swizzles (inverse of ISA 7.12.2 wave32 layouts).
// A 16x32 subtile, element (rr,k) lives at lane*16+elem with:
__device__ __forceinline__ int a_slot(int rr, int k) {
    int lane = rr + (((k >> 3) & 1) << 4);
    int e    = (k & 7) + ((k >> 4) << 3);
    return lane * 16 + e;
}
// B 32x16 subtile, element (k,n):
__device__ __forceinline__ int b_slot(int k, int n) {
    int lane = n + (((k >> 3) & 1) << 4);
    int e    = (k & 7) + ((k >> 4) << 3);
    return lane * 16 + e;
}

// ---------------- table builder: swizzled bf16 constant operands ----------------
// FwSw: forward W-DFT matrix [k=0..255][n=0..31] (B-fragment order, [kchunk(8)][nsub(2)][512])
// GSw : inverse W-DFT matrix [k=0..31][w=0..255] with irfft Hermitian weights
//       folded in ([wsub(16)][512])
// WskSw: skip weights [L][kchunk(2)][osub(4)][512] in A-fragment order.
__global__ void __launch_bounds__(256) fno_build_tables(const float* __restrict__ skip_w,
                                                        bf16* __restrict__ FwSw,
                                                        bf16* __restrict__ GSw,
                                                        bf16* __restrict__ WskSw) {
    int i = blockIdx.x * 256 + threadIdx.x;
    if (i < 256 * 32) {                       // FwSw
        int k = i >> 5, n = i & 31;
        float v = 0.f;
        if (n < 24) {
            int j = n >> 1;
            float th = -2.f * PI_F * (float)j * (float)k / 256.f;
            v = (n & 1) ? sinf(th) : cosf(th);
        }
        FwSw[(k >> 5) * 1024 + (n >> 4) * 512 + b_slot(k & 31, n & 15)] = f2bf(v);
    }
    if (i < 32 * 256) {                       // GSw
        int k = i >> 8, w = i & 255;
        float v = 0.f;
        if (k < 24) {
            int j = k >> 1;
            float cn = (j == 0) ? 1.f : 2.f;
            float sc = cn / (float)HWP;
            float th = 2.f * PI_F * (float)j * (float)w / 256.f;
            v = (k & 1) ? (-sinf(th) * sc) : (cosf(th) * sc);
        }
        GSw[(w >> 4) * 512 + b_slot(k, w & 15)] = f2bf(v);
    }
    if (i < NL * 64 * 64) {                   // WskSw
        int l = i >> 12, o = (i >> 6) & 63, c = i & 63;
        float v = skip_w[(size_t)l * 4096 + o * 64 + c];
        WskSw[l * 4096 + (c >> 5) * 2048 + (o >> 4) * 512 + a_slot(o & 15, c & 31)] = f2bf(v);
    }
}

// ---------------- lift: h[b,c,p] = bf16( x[b,p]*w[c] + bias[c] ) ----------------
__global__ void __launch_bounds__(256) fno_lift(const float* __restrict__ x,
                                                const float* __restrict__ w,
                                                const float* __restrict__ bias,
                                                bf16* __restrict__ h) {
    size_t i = (size_t)blockIdx.x * 256 + threadIdx.x;
    if (i >= (size_t)NB * NC * HWP) return;
    size_t p = i % HWP;
    int c = (int)((i / HWP) % NC);
    int b = (int)(i / ((size_t)HWP * NC));
    h[i] = f2bf(x[(size_t)b * HWP + p] * w[c] + bias[c]);
}

// ---------------- skip conv: out[b,o,p] = sum_c w[o,c] h[b,c,p] + bias[o] ----
// M=64 x N=128 block tile, K=64 in two 32-chunks. Weights come pre-swizzled
// from global (wsw); bf16 activations staged raw into fragment-order LDS.
__global__ void __launch_bounds__(256) fno_skip_gemm(const bf16* __restrict__ h,
                                                     const bf16* __restrict__ wsw,
                                                     const float* __restrict__ bias,
                                                     bf16* __restrict__ out) {
    __shared__ alignas(32) bf16 bSw[8 * 512];   // 8 col-subtiles of 32x16
    int b    = blockIdx.y;
    int p0   = blockIdx.x * 128;
    int tid  = threadIdx.x;
    int wave = tid >> 5;
    int lane = tid & 31;
    int mi   = wave & 3;            // A subtile (rows mi*16..)
    int cq   = (wave >> 2) * 4;     // first B subtile (cols cq*16..)
    const bf16* hb = h + (size_t)b * NC * HWP;
    v8f acc[4] = {v8f{}, v8f{}, v8f{}, v8f{}};

    for (int kc = 0; kc < 2; ++kc) {
        // stage 32x128 activation chunk: raw bf16 pairs along k, packed b32 stores
        for (int p = tid; p < 2048; p += 256) {
            int kk = p >> 7, n = p & 127;       // n fastest -> coalesced
            int k  = kk * 2;
            bf16 lo = hb[(size_t)(kc * 32 + k)     * HWP + p0 + n];
            bf16 hi = hb[(size_t)(kc * 32 + k + 1) * HWP + p0 + n];
            *(unsigned*)&bSw[(n >> 4) * 512 + b_slot(k, n & 15)] = pack2b(lo, hi);
        }
        if (kc == 0)
            __builtin_prefetch(hb + (size_t)32 * HWP + p0 + (tid & 127), 0, 0);
        __syncthreads();
        v16bf af = *(const v16bf*)(wsw + kc * 2048 + mi * 512 + lane * 16);
#pragma unroll
        for (int j = 0; j < 4; ++j) {
            v16bf bfr = *(const v16bf*)(&bSw[(cq + j) * 512 + lane * 16]);
            acc[j] = wmma_bf16(af, bfr, acc[j]);
        }
        __syncthreads();
    }
#pragma unroll
    for (int j4 = 0; j4 < 4; ++j4) {
        int col = p0 + (cq + j4) * 16 + (lane & 15);
#pragma unroll
        for (int j = 0; j < 8; ++j) {
            int o = mi * 16 + 8 * (lane >> 4) + j;
            out[(size_t)b * NC * HWP + (size_t)o * HWP + col] = f2bf(acc[j4][j] + bias[o]);
        }
    }
}

// ---------------- forward W-axis truncated DFT ----------------
// U[r, 2n|2n+1] = sum_w h[r,w] * {cos,-sin}(2pi n w/256), r=(b,c,hh), n<12.
// GEMM [NR x 256] x [256 x 24(pad32)]. Block: 128 rows, wave: 16 rows x 32 cols.
// A staging is a pure b32 copy (bf16 activations); DFT fragments from global.
__global__ void __launch_bounds__(256) fno_fwd_wdft(const bf16* __restrict__ h,
                                                    const bf16* __restrict__ FwSw,
                                                    float* __restrict__ U) {
    __shared__ alignas(32) bf16 aSw[8 * 512];   // 8 row-subtiles of 16x32
    int r0   = blockIdx.x * 128;
    int tid  = threadIdx.x;
    int wave = tid >> 5;
    int lane = tid & 31;
    v8f acc0 = v8f{}, acc1 = v8f{};

    for (int kc = 0; kc < 8; ++kc) {
        int k0 = kc * 32;
        // stage 128x32 chunk of h: aligned u32 loads (2 bf16), u32 LDS stores
        for (int p = tid; p < 2048; p += 256) {
            int r = p >> 4, kk = p & 15;        // kk fastest -> coalesced
            unsigned v = *(const unsigned*)(h + (size_t)(r0 + r) * 256 + k0 + kk * 2);
            *(unsigned*)&aSw[(r >> 4) * 512 + a_slot(r & 15, kk * 2)] = v;
        }
        if (kc < 7)
            __builtin_prefetch(h + (size_t)(r0 + (tid >> 1)) * 256 + k0 + 32, 0, 0);
        __syncthreads();
        v16bf af  = *(const v16bf*)(&aSw[wave * 512 + lane * 16]);
        v16bf bf0 = *(const v16bf*)(FwSw + kc * 1024 + lane * 16);
        v16bf bf1 = *(const v16bf*)(FwSw + kc * 1024 + 512 + lane * 16);
        acc0 = wmma_bf16(af, bf0, acc0);
        acc1 = wmma_bf16(af, bf1, acc1);
        __syncthreads();
    }
    int coln = lane & 15;
#pragma unroll
    for (int j = 0; j < 8; ++j) {
        int rr = r0 + wave * 16 + 8 * (lane >> 4) + j;
        U[(size_t)rr * 24 + coln] = acc0[j];                     // cols 0..15
        if (coln < 8) U[(size_t)rr * 24 + 16 + coln] = acc1[j];  // cols 16..23
    }
}

// ---------------- forward H-axis DFT to 24 modes (small, VALU) ----------------
__global__ void __launch_bounds__(256) fno_fwd_hdft(const float* __restrict__ U,
                                                    float* __restrict__ V) {
    int i = blockIdx.x * 256 + threadIdx.x;       // (bc, m, n)
    if (i >= NBC * 24 * NK) return;
    int n = i % NK, m = (i / NK) % 24, bc = i / (NK * 24);
    int f = (m < 12) ? m : (m - 24);
    float w0 = -2.f * PI_F * (float)f / 256.f;
    const float* u = U + (size_t)bc * 256 * 24 + 2 * n;
    float re = 0.f, im = 0.f;
    for (int hh = 0; hh < 256; ++hh) {
        float s, c;
        __sincosf(w0 * (float)hh, &s, &c);
        float a = u[hh * 24], d = u[hh * 24 + 1];
        re += a * c - d * s;
        im += a * s + d * c;
    }
    V[(size_t)i * 2]     = re;
    V[(size_t)i * 2 + 1] = im;
}

// ---------------- complex channel mixing in mode space (small) ----------------
__global__ void __launch_bounds__(256) fno_mode_mix(const float* __restrict__ V,
                                                    const float* __restrict__ w1re,
                                                    const float* __restrict__ w1im,
                                                    const float* __restrict__ w2re,
                                                    const float* __restrict__ w2im,
                                                    float* __restrict__ V2) {
    int i = blockIdx.x * 256 + threadIdx.x;       // (b, o, m, n)
    if (i >= NB * NC * 24 * NK) return;
    int n = i % NK, m = (i / NK) % 24, o = (i / (NK * 24)) % NC, b = i / (NK * 24 * NC);
    const float* wre; const float* wim; int mm;
    if (m < 12) { wre = w1re; wim = w1im; mm = m; }
    else        { wre = w2re; wim = w2im; mm = m - 12; }
    float re = 0.f, im = 0.f;
    for (int c = 0; c < NC; ++c) {
        size_t vi = (((size_t)(b * NC + c) * 24 + m) * NK + n) * 2;
        float a = V[vi], d = V[vi + 1];
        size_t wi = ((size_t)(c * NC + o)) * (NK * NK) + mm * NK + n;
        float wr = wre[wi], wj = wim[wi];
        re += a * wr - d * wj;
        im += a * wj + d * wr;
    }
    size_t oi = (((size_t)(b * NC + o) * 24 + m) * NK + n) * 2;
    V2[oi] = re; V2[oi + 1] = im;
}

// ---------------- inverse H-axis DFT (small, VALU) ----------------
__global__ void __launch_bounds__(256) fno_inv_hdft(const float* __restrict__ V2,
                                                    float* __restrict__ T) {
    int i = blockIdx.x * 256 + threadIdx.x;       // (bc, hh, n)
    if (i >= NBC * 256 * NK) return;
    int n = i % NK, hh = (i / NK) & 255, bc = i / (NK * 256);
    float re = 0.f, im = 0.f;
    for (int m = 0; m < 24; ++m) {
        int f = (m < 12) ? m : (m - 24);
        float s, c;
        __sincosf(2.f * PI_F * (float)f * (float)hh / 256.f, &s, &c);
        size_t vi = (((size_t)bc * 24 + m) * NK + n) * 2;
        float a = V2[vi], d = V2[vi + 1];
        re += a * c - d * s;
        im += a * s + d * c;
    }
    size_t ti = ((size_t)bc * 256 + hh) * 24 + 2 * n;
    T[ti] = re; T[ti + 1] = im;
}

// ---------------- inverse W-axis DFT + skip add + exact GELU ----------------
// hb[r,w] = bf16( gelu( sum_k T[r,k] * G[k,w] + hb[r,w] ) ), K=24 padded to 32.
__global__ void __launch_bounds__(256) fno_inv_wdft_gelu(const float* __restrict__ T,
                                                         const bf16* __restrict__ GSw,
                                                         bf16* __restrict__ hb) {
    __shared__ alignas(32) bf16 aSw[2 * 512];
    int r0   = blockIdx.x * 32;
    int tid  = threadIdx.x;
    int wave = tid >> 5;
    int lane = tid & 31;

    // stage 32 rows x 32 K (zero-pad K>=24): 512 pairs
    for (int p = tid; p < 512; p += 256) {
        int r = p >> 4, kk = p & 15;
        unsigned pk = 0;
        if (kk < 12) {
            float2 v = *(const float2*)(T + (size_t)(r0 + r) * 24 + kk * 2);
            pk = pack2f(v.x, v.y);
        }
        *(unsigned*)&aSw[(r >> 4) * 512 + a_slot(r & 15, kk * 2)] = pk;
    }
    __syncthreads();

    int mi = wave >> 2;           // 0..1  (16-row tile)
    int c0 = (wave & 3) * 4;      // first B subtile
    v16bf af = *(const v16bf*)(&aSw[mi * 512 + lane * 16]);
    v8f acc[4] = {v8f{}, v8f{}, v8f{}, v8f{}};
#pragma unroll
    for (int j4 = 0; j4 < 4; ++j4) {
        v16bf bfr = *(const v16bf*)(GSw + (c0 + j4) * 512 + lane * 16);
        acc[j4] = wmma_bf16(af, bfr, acc[j4]);
    }
#pragma unroll
    for (int j4 = 0; j4 < 4; ++j4) {
        int col = (c0 + j4) * 16 + (lane & 15);
#pragma unroll
        for (int j = 0; j < 8; ++j) {
            int rr = r0 + mi * 16 + 8 * (lane >> 4) + j;
            size_t idx = (size_t)rr * 256 + col;
            float v = acc[j4][j] + bf2f(hb[idx]);
            hb[idx] = f2bf(0.5f * v * (1.0f + erff(v * 0.70710678118654752f)));
        }
    }
}

// ---------------- projection: out[b,p] = sum_c w[c] h[b,c,p] + bias ----------
__global__ void __launch_bounds__(256) fno_proj(const bf16* __restrict__ h,
                                                const float* __restrict__ w,
                                                const float* __restrict__ bias,
                                                float* __restrict__ out) {
    size_t i = (size_t)blockIdx.x * 256 + threadIdx.x;   // (b, p)
    if (i >= (size_t)NB * HWP) return;
    size_t p = i % HWP;
    int b = (int)(i / HWP);
    const bf16* hbp = h + (size_t)b * NC * HWP + p;
    float acc = bias[0];
#pragma unroll 4
    for (int c = 0; c < NC; ++c) acc += bf2f(hbp[(size_t)c * HWP]) * w[c];
    out[i] = acc;
}

// ============================ launcher ============================
extern "C" void kernel_launch(void* const* d_in, const int* in_sizes, int n_in,
                              void* d_out, int out_size, void* d_ws, size_t ws_size,
                              hipStream_t stream) {
    const float* x        = (const float*)d_in[0];
    const float* lift_w   = (const float*)d_in[1];
    const float* lift_b   = (const float*)d_in[2];
    const float* spec1_re = (const float*)d_in[3];
    const float* spec1_im = (const float*)d_in[4];
    const float* spec2_re = (const float*)d_in[5];
    const float* spec2_im = (const float*)d_in[6];
    const float* skip_w   = (const float*)d_in[7];
    const float* skip_b   = (const float*)d_in[8];
    const float* proj_w   = (const float*)d_in[9];
    const float* proj_b   = (const float*)d_in[10];
    float* out = (float*)d_out;

    // workspace layout
    const size_t SZ_H = (size_t)NB * NC * HWP;        // 67,108,864 elems (bf16)
    const size_t SZ_U = (size_t)NBC * 256 * 24;       //  6,291,456 f32 (reused as T)
    const size_t SZ_V = (size_t)NBC * 24 * NK * 2;    //    589,824 f32
    bf16*  hA = (bf16*)d_ws;
    bf16*  hB = hA + SZ_H;
    float* U  = (float*)(hB + SZ_H);
    float* V  = U + SZ_U;
    float* V2 = V + SZ_V;
    bf16*  FwSw  = (bf16*)(V2 + SZ_V);                // 8 * 1024
    bf16*  GSw   = FwSw + 8 * 1024;                   // 16 * 512
    bf16*  WskSw = GSw + 16 * 512;                    // NL * 4096

    const size_t specL = (size_t)NC * NC * NK * NK;   // per-layer spectral weights

    fno_build_tables<<<64, 256, 0, stream>>>(skip_w, FwSw, GSw, WskSw);

    fno_lift<<<(unsigned)((SZ_H + 255) / 256), 256, 0, stream>>>(x, lift_w, lift_b, hA);

    bf16* cur = hA;
    bf16* nxt = hB;
    for (int l = 0; l < NL; ++l) {
        fno_skip_gemm<<<dim3(HWP / 128, NB), 256, 0, stream>>>(
            cur, WskSw + (size_t)l * 4096, skip_b + (size_t)l * NC, nxt);
        fno_fwd_wdft<<<NR / 128, 256, 0, stream>>>(cur, FwSw, U);
        fno_fwd_hdft<<<(NBC * 24 * NK + 255) / 256, 256, 0, stream>>>(U, V);
        fno_mode_mix<<<(NB * NC * 24 * NK + 255) / 256, 256, 0, stream>>>(
            V, spec1_re + (size_t)l * specL, spec1_im + (size_t)l * specL,
               spec2_re + (size_t)l * specL, spec2_im + (size_t)l * specL, V2);
        fno_inv_hdft<<<(NBC * 256 * NK + 255) / 256, 256, 0, stream>>>(V2, U /* as T */);
        fno_inv_wdft_gelu<<<NR / 32, 256, 0, stream>>>(U, GSw, nxt);
        bf16* t = cur; cur = nxt; nxt = t;
    }

    fno_proj<<<(unsigned)(((size_t)NB * HWP + 255) / 256), 256, 0, stream>>>(
        cur, proj_w, proj_b, out);
}
